// idctmodule2D_27315992003322
// MI455X (gfx1250) — matrix operand order repaired and checked by reading, not compiled, not run on hardware
//
#include <hip/hip_runtime.h>
#include <hip/hip_bf16.h>

// 2D orthonormal IDCT (DCT-III) over (32,3,512,512) as two f16 WMMA GEMMs
// per image with f32 accumulation, fragment-native memory layouts so no
// transpose/shuffle work is ever needed.

typedef __attribute__((ext_vector_type(16))) _Float16 v16h;
typedef __attribute__((ext_vector_type(8)))  _Float16 v8h;
typedef __attribute__((ext_vector_type(8)))  float    v8f;

#define NDIM   512
#define IMG_EL (512 * 512)   // 262144 elements per image
#define NIMG   96            // 32 * 3

// c_k * cos(pi * k * (2m+1) / 1024), ortho DCT-III basis for N=512.
// Exact integer phase reduction (period 2048 in units of pi/1024) keeps
// cosf arguments in [0, 2pi) for full float accuracy.
__device__ __forceinline__ float idct_basis(int k, int m) {
  int phase = (k * (2 * m + 1)) & 2047;
  float c = (k == 0) ? 0.04419417382415922f   // 1/sqrt(512)
                     : 0.0625f;               // sqrt(2/512)
  return c * cosf((float)phase * 0.0030679615757712823f); // pi/1024
}

// ---- A-fragment layout (16-bit A, 16x32, contiguous k order) --------------------
// addr(halfs) = ((mb*16 + kb)*32 + lane)*16 + j
//   lane<16 : j=0..15 -> K=kb*32+j
//   lane>=16: j=0..15 -> K=kb*32+16+j
__device__ __forceinline__ void a_decode(int hidx, int& m, int& k) {
  int j  = hidx & 15;
  int l  = (hidx >> 4) & 31;
  int kb = (hidx >> 9) & 15;   // K/32, 16 blocks
  int mb = hidx >> 13;         // M/16, 32 blocks
  m = mb * 16 + (l & 15);
  int kk = (l < 16) ? j : j + 16;
  k = kb * 32 + kk;
}

// ---- B-fragment layout (16-bit B, 32x16 slabs) --------------------------
// element (k,n): addr(halfs) = ((nb*32 + kb16)*256) + nr*16 + kr
//   kb16=k/16, kr=k&15, nb=n/16, nr=n&15
// A wave's B fragment for K-block t is then simply base + lane*16, 16
// contiguous halfs per lane (lanes 0-15: K t*32..+15, lanes 16-31: +16..31).

__global__ __launch_bounds__(256) void build_mh_kernel(_Float16* __restrict__ Mh) {
  int idx = blockIdx.x * blockDim.x + threadIdx.x;   // 0..262143
  int m, k;
  a_decode(idx, m, k);
  Mh[idx] = (_Float16)idct_basis(k, m);              // Mh[m][k] = c_k cos(pi k (2m+1)/1024)
}

__global__ __launch_bounds__(256) void build_mwt_kernel(_Float16* __restrict__ MwT) {
  int idx = blockIdx.x * blockDim.x + threadIdx.x;   // 0..262143
  int kr   = idx & 15;
  int nr   = (idx >> 4) & 15;
  int kb16 = (idx >> 8) & 31;
  int nb   = idx >> 13;
  int k = kb16 * 16 + kr;
  int n = nb * 16 + nr;
  MwT[idx] = (_Float16)idct_basis(k, n);             // MwT[k][n] = Mw[n][k]
}

__global__ __launch_bounds__(256) void convert_x_kernel(const float* __restrict__ X,
                                                        _Float16* __restrict__ Xh) {
  int idx  = blockIdx.x * blockDim.x + threadIdx.x;  // over 96*262144 halfs
  int hidx = idx & (IMG_EL - 1);
  int img  = idx >> 18;
  int m, k;
  a_decode(hidx, m, k);
  Xh[idx] = (_Float16)X[(size_t)img * IMG_EL + m * NDIM + k];
}

// ---- GEMM: C[512x512] = A[512x512] * B[512x512], f16 in / f32 acc -------
// 256 threads = 8 waves. Wave grid 4(M) x 2(N); each wave owns a 32x64 tile
// (2x4 fragments). Workgroup tile 128x128, grid.x = 16 tiles, grid.y = image.
// OUT_F32 == 0: store f16 fragments in B-fragment layout (feeds pass 2).
// OUT_F32 == 1: store f32 row-major (final output).
template <int OUT_F32>
__global__ __launch_bounds__(256) void gemm512_wmma(const _Float16* __restrict__ A,
                                                    size_t aStride,
                                                    const _Float16* __restrict__ B,
                                                    size_t bStride,
                                                    void* __restrict__ Dp,
                                                    size_t dStride) {
  const int img  = blockIdx.y;
  const int bx   = blockIdx.x & 3;    // N macro-tile
  const int by   = blockIdx.x >> 2;   // M macro-tile
  const int lane = threadIdx.x & 31;
  const int wave = threadIdx.x >> 5;
  const int wm   = wave & 3;
  const int wn   = wave >> 2;

  const _Float16* Ai = A + (size_t)img * aStride;
  const _Float16* Bi = B + (size_t)img * bStride;

  const int mb0 = by * 8 + wm * 2;    // 16-row fragment blocks (2 of them)
  const int nb0 = bx * 8 + wn * 4;    // 16-col fragment blocks (4 of them)

  v8f acc[2][4] = {};

  for (int t = 0; t < 16; ++t) {      // K blocks of 32
    v16h a[2], b[4];
#pragma unroll
    for (int i = 0; i < 2; ++i) {
      const _Float16* p = Ai + ((size_t)((mb0 + i) * 16 + t) * 32 + lane) * 16;
      a[i] = *(const v16h*)p;
    }
#pragma unroll
    for (int j = 0; j < 4; ++j) {
      const _Float16* p = Bi + (size_t)((nb0 + j) * 32 + 2 * t) * 256 + (size_t)lane * 16;
      b[j] = *(const v16h*)p;
    }
    if (t + 1 < 16) {                 // global_prefetch_b8 for next K slab
      __builtin_prefetch(Ai + ((size_t)(mb0 * 16 + t + 1) * 32 + lane) * 16, 0, 1);
      __builtin_prefetch(Bi + (size_t)(nb0 * 32 + 2 * (t + 1)) * 256 + (size_t)lane * 16, 0, 1);
    }
#pragma unroll
    for (int i = 0; i < 2; ++i)
#pragma unroll
      for (int j = 0; j < 4; ++j)
        acc[i][j] = __builtin_amdgcn_wmma_f32_16x16x32_f16(
            /*neg_a=*/false, a[i], /*neg_b=*/false, b[j],
            /*c_mod=*/(short)0, acc[i][j],
            /*reuse_a=*/false, /*reuse_b=*/false);
  }

  if (OUT_F32) {
    // f32 C/D layout: lanes 0-15 hold M=0..7 (N=lane), lanes 16-31 M=8..15.
    float* D = (float*)Dp + (size_t)img * dStride;
    const int nloc  = lane & 15;
    const int mHalf = (lane < 16) ? 0 : 8;
#pragma unroll
    for (int i = 0; i < 2; ++i)
#pragma unroll
      for (int j = 0; j < 4; ++j) {
        const int n = (nb0 + j) * 16 + nloc;
        const int m = (mb0 + i) * 16 + mHalf;
#pragma unroll
        for (int r = 0; r < 8; ++r)
          D[(size_t)(m + r) * NDIM + n] = acc[i][j][r];
      }
  } else {
    // Convert to f16 and drop straight into pass-2's B-fragment layout:
    // one b128 store per lane per fragment.
    _Float16* D = (_Float16*)Dp + (size_t)img * dStride;
#pragma unroll
    for (int i = 0; i < 2; ++i)
#pragma unroll
      for (int j = 0; j < 4; ++j) {
        size_t blockBase = (size_t)((nb0 + j) * 32 + (mb0 + i)) * 256;
        size_t off = blockBase + ((lane < 16) ? (size_t)lane * 16
                                              : (size_t)(lane - 16) * 16 + 8);
        v8h h;
#pragma unroll
        for (int r = 0; r < 8; ++r) h[r] = (_Float16)acc[i][j][r];
        *(v8h*)(D + off) = h;
      }
  }
}

extern "C" void kernel_launch(void* const* d_in, const int* in_sizes, int n_in,
                              void* d_out, int out_size, void* d_ws, size_t ws_size,
                              hipStream_t stream) {
  const float* X = (const float*)d_in[0];  // (32,3,512,512) f32
  // W_r/W_i inputs are not needed: the basis is recomputed exactly on device.

  _Float16* ws  = (_Float16*)d_ws;
  _Float16* Xh  = ws;                                   // 96*262144 halfs (~50 MB)
  _Float16* Y1  = ws + (size_t)NIMG * IMG_EL;           // 96*262144 halfs (~50 MB)
  _Float16* MwT = ws + (size_t)2 * NIMG * IMG_EL;       // 262144 halfs
  _Float16* Mh  = MwT + IMG_EL;                         // 262144 halfs
  float* out = (float*)d_out;

  build_mwt_kernel<<<IMG_EL / 256, 256, 0, stream>>>(MwT);
  build_mh_kernel<<<IMG_EL / 256, 256, 0, stream>>>(Mh);
  convert_x_kernel<<<(NIMG * IMG_EL) / 256, 256, 0, stream>>>(X, Xh);

  // Pass 1 (IDCT over width):  Y1 = X_img * MwT   -> f16 fragment layout
  gemm512_wmma<0><<<dim3(16, NIMG), 256, 0, stream>>>(
      Xh, (size_t)IMG_EL, MwT, (size_t)0, (void*)Y1, (size_t)IMG_EL);

  // Pass 2 (IDCT over height): out = Mh * Y1      -> f32 row-major
  gemm512_wmma<1><<<dim3(16, NIMG), 256, 0, stream>>>(
      Mh, (size_t)0, Y1, (size_t)IMG_EL, (void*)out, (size_t)IMG_EL);
}
